// BiLSTM_9474697855555
// MI455X (gfx1250) — matrix operand order repaired and checked by reading, not compiled, not run on hardware
//
#include <hip/hip_runtime.h>
#include <hip/hip_bf16.h>

typedef __bf16 bf16_t;
typedef __attribute__((ext_vector_type(16))) __bf16 v16bf;
typedef __attribute__((ext_vector_type(8)))  float  v8f;
typedef __attribute__((ext_vector_type(4)))  int    v4i;

#define B_SZ 64
#define S_SZ 512
#define I_SZ 256
#define H_SZ 512
#define NG   2048   /* 4*H */
#define KTOT 768    /* I+H */
#define NBLOCKS 32u

#define LDS_A_BYTES (B_SZ * KTOT * 2)            /* 98304: A = [x_t|h], 64x768 bf16 */
#define LDS_G_BYTES (8 * B_SZ * 16 * 4)          /* 32768: per-wave 64x16 f32 gate scratch */
#define LDS_TOTAL   (LDS_A_BYTES + LDS_G_BYTES)  /* 131072 */

#if __has_builtin(__builtin_amdgcn_global_load_async_to_lds_b128)
#define HAS_ASYNC_LDS 1
#else
#define HAS_ASYNC_LDS 0
#endif

// ---- prep: f32 -> bf16 input conversion -------------------------------------
__global__ __launch_bounds__(256) void prep_x(const float* __restrict__ x,
                                              bf16_t* __restrict__ xbf) {
  size_t idx = (size_t)blockIdx.x * 256 + threadIdx.x;   // B*S*I elements
  xbf[idx] = (bf16_t)x[idx];
}

// ---- prep: transpose + gate-interleave weights ------------------------------
// Permuted column n' = 4*j + gate  <->  original column gate*512 + j.
__global__ __launch_bounds__(256) void prep_w(const float* __restrict__ Wf,
                                              const float* __restrict__ Wb,
                                              bf16_t* __restrict__ wt) {
  size_t idx = (size_t)blockIdx.x * 256 + threadIdx.x;   // 2*KTOT*NG elements
  int dir = (int)(idx / ((size_t)KTOT * NG));
  size_t r = idx % ((size_t)KTOT * NG);
  int k = (int)(r / NG);
  int n = (int)(r % NG);                                  // permuted column index
  int orig = (n & 3) * H_SZ + (n >> 2);
  const float* src = dir ? Wb : Wf;
  wt[((size_t)dir * NG + n) * KTOT + k] = (bf16_t)src[(size_t)k * NG + orig];
}

// ---- zero h ping buffer + c + barrier counters ------------------------------
__global__ __launch_bounds__(256) void zero_ws(uint4* __restrict__ p, int n16) {
  int idx = blockIdx.x * 256 + threadIdx.x;
  if (idx < n16) p[idx] = make_uint4(0u, 0u, 0u, 0u);
}

// ---- persistent kernel: all 512 steps, grid barrier between steps -----------
// grid: 32 blocks x 256 threads (8 waves). block -> (dir, 8 n-tiles).
__global__ __launch_bounds__(256) void lstm_persist(
    const bf16_t* __restrict__ xbf, const bf16_t* __restrict__ wt,
    const float* __restrict__ biasf, const float* __restrict__ biasb,
    bf16_t* __restrict__ hbuf0, bf16_t* __restrict__ hbuf1,
    float* __restrict__ cst, unsigned* __restrict__ ctr,
    float* __restrict__ out)
{
  extern __shared__ char smem[];
  bf16_t* Abuf = (bf16_t*)smem;                 // [64][768] bf16
  float*  gsc  = (float*)(smem + LDS_A_BYTES);  // [8 waves][64][16] f32

  const int tid   = threadIdx.x;
  const int lane  = tid & 31;
  const int wid   = tid >> 5;
  const int dir   = blockIdx.x >> 4;                 // 16 blocks per direction
  const int ntile = (blockIdx.x & 15) * 8 + wid;     // 0..127
  const int lo    = lane & 15;
  const int hi    = lane >> 4;
  const int np    = ntile * 16 + lo;                 // permuted column
  const int jbase = ntile * 4;

  const float bv = (dir ? biasb : biasf)[(np & 3) * H_SZ + (np >> 2)];
  const bf16_t* wrow = wt + ((size_t)dir * NG + np) * KTOT;

  for (int s = 0; s < S_SZ; ++s) {
    const bf16_t* hin  = (s & 1) ? hbuf1 : hbuf0;
    bf16_t*       hout = (s & 1) ? hbuf0 : hbuf1;
    const int t = dir ? (S_SZ - 1 - s) : s;

    // ---- stage A = [x_t | h] into LDS ----
    const bf16_t* hrow = hin + (size_t)dir * B_SZ * H_SZ;
    #pragma unroll
    for (int c = tid; c < B_SZ * (KTOT / 8); c += 256) {   // 24 iters, 16B each
      int row = c / (KTOT / 8);
      int kc  = (c % (KTOT / 8)) * 8;
      bf16_t* dst = &Abuf[row * KTOT + kc];
#if HAS_ASYNC_LDS
      if (kc < I_SZ) {
        // x is read-only: default (WGP) scope, cache-friendly.
        __builtin_amdgcn_global_load_async_to_lds_b128(
            (__attribute__((address_space(1))) v4i*)(xbf + ((size_t)row * S_SZ + t) * I_SZ + kc),
            (__attribute__((address_space(3))) v4i*)dst, 0, /*cpol=*/0);
      } else {
        // h was written by OTHER blocks last step: scope DEV bypasses WGP$.
        __builtin_amdgcn_global_load_async_to_lds_b128(
            (__attribute__((address_space(1))) v4i*)(hrow + (size_t)row * H_SZ + (kc - I_SZ)),
            (__attribute__((address_space(3))) v4i*)dst, 0, /*cpol=*/0x10);
      }
#else
      const bf16_t* src = (kc < I_SZ)
          ? (xbf + ((size_t)row * S_SZ + t) * I_SZ + kc)
          : (hrow + (size_t)row * H_SZ + (kc - I_SZ));
      *(float4*)dst = *(const float4*)src;
#endif
    }
#if HAS_ASYNC_LDS
#if __has_builtin(__builtin_amdgcn_s_wait_asynccnt)
    __builtin_amdgcn_s_wait_asynccnt(0);
#else
    asm volatile("s_wait_asynccnt 0x0" ::: "memory");
#endif
#endif
    __syncthreads();

    // ---- GEMM: D(64x16) = A(64x768) @ Wt-column-block + bias ----
    v8f acc[4];
    #pragma unroll
    for (int m = 0; m < 4; ++m)
      #pragma unroll
      for (int i = 0; i < 8; ++i) acc[m][i] = bv;

    #pragma unroll 2
    for (int k0 = 0; k0 < KTOT; k0 += 32) {
      const bf16_t* bp = wrow + k0 + hi * 8;
      v16bf bfrag;
      ((float4*)&bfrag)[0] = *(const float4*)bp;
      ((float4*)&bfrag)[1] = *(const float4*)(bp + 16);
      v16bf af[4];
      #pragma unroll
      for (int m = 0; m < 4; ++m) {
        const bf16_t* ap = &Abuf[(m * 16 + lo) * KTOT + k0 + hi * 8];
        ((float4*)&af[m])[0] = *(const float4*)ap;
        ((float4*)&af[m])[1] = *(const float4*)(ap + 16);
      }
      #pragma unroll
      for (int m = 0; m < 4; ++m)
        acc[m] = __builtin_amdgcn_wmma_f32_16x16x32_bf16(
            false, af[m], false, bfrag, (short)0, acc[m], false, false);
    }

    // ---- spill gate tile to per-wave LDS scratch (cross-lane regroup) ----
    float* gw = gsc + wid * (B_SZ * 16);
    #pragma unroll
    for (int m = 0; m < 4; ++m)
      #pragma unroll
      for (int v = 0; v < 8; ++v)
        gw[(m * 16 + hi * 8 + v) * 16 + lo] = acc[m][v];
    __syncthreads();

    // ---- LSTM cell update: this wave owns 4 hidden units x 64 batches ----
    #pragma unroll
    for (int it = 0; it < 8; ++it) {
      int item = it * 32 + lane;        // 256 items = 64 b x 4 jj
      int b  = item >> 2;
      int jj = item & 3;
      int j  = jbase + jj;
      float4 g4 = *(const float4*)&gw[b * 16 + jj * 4];  // f, i, o, g
      size_t ci = (size_t)dir * B_SZ * H_SZ + (size_t)b * H_SZ + j;
      float c  = cst[ci];                     // block-private: cacheable
      float sf = 1.f / (1.f + __expf(-g4.x));
      float si = 1.f / (1.f + __expf(-g4.y));
      float so = 1.f / (1.f + __expf(-g4.z));
      float cn = sf * c + si * tanhf(g4.w);
      float hn = so * tanhf(cn);
      cst[ci]  = cn;
      hout[ci] = (bf16_t)hn;
      size_t oi = ((size_t)b * S_SZ + t) * H_SZ + j;
      float v = 0.5f * hn;
      if (s < S_SZ / 2) {
        out[oi] = v;                          // first writer of this location
      } else {
        // second writer: first write came from the other direction's block
        float prev = __hip_atomic_load(&out[oi], __ATOMIC_RELAXED,
                                       __HIP_MEMORY_SCOPE_AGENT);
        out[oi] = prev + v;
      }
    }

    // ---- device-scope grid barrier (per-step counter, no reset races) ----
    __syncthreads();
    __threadfence();                          // flush h/out to device scope
    if (tid == 0) {
      __hip_atomic_fetch_add(&ctr[s], 1u, __ATOMIC_ACQ_REL,
                             __HIP_MEMORY_SCOPE_AGENT);
      while (__hip_atomic_load(&ctr[s], __ATOMIC_ACQUIRE,
                               __HIP_MEMORY_SCOPE_AGENT) < NBLOCKS)
        __builtin_amdgcn_s_sleep(2);
    }
    __syncthreads();
  }
}

extern "C" void kernel_launch(void* const* d_in, const int* in_sizes, int n_in,
                              void* d_out, int out_size, void* d_ws, size_t ws_size,
                              hipStream_t stream) {
  (void)in_sizes; (void)n_in; (void)out_size; (void)ws_size;
  const float* x   = (const float*)d_in[0];
  const float* Wf  = (const float*)d_in[1];
  const float* bf_ = (const float*)d_in[2];
  const float* Wb  = (const float*)d_in[3];
  const float* bb_ = (const float*)d_in[4];
  float* out = (float*)d_out;

  char* w = (char*)d_ws;
  bf16_t*   xbf  = (bf16_t*)w;   w += (size_t)B_SZ * S_SZ * I_SZ * 2;  // 16 MB
  bf16_t*   wt   = (bf16_t*)w;   w += (size_t)2 * NG * KTOT * 2;       //  6 MB
  bf16_t*   hbf0 = (bf16_t*)w;   w += (size_t)2 * B_SZ * H_SZ * 2;     // 128 KB (zeroed)
  float*    cst  = (float*)w;    w += (size_t)2 * B_SZ * H_SZ * 4;     // 256 KB (zeroed)
  unsigned* ctr  = (unsigned*)w; w += (size_t)S_SZ * 4;                //   2 KB (zeroed)
  bf16_t*   hbf1 = (bf16_t*)w;                                         // 128 KB (ping)

  prep_x<<<(B_SZ * S_SZ * I_SZ) / 256, 256, 0, stream>>>(x, xbf);
  prep_w<<<(2 * KTOT * NG) / 256, 256, 0, stream>>>(Wf, Wb, wt);
  // hbf0 + cst + ctr adjacent: 131072 + 262144 + 2048 = 395264 B = 24704 uint4
  zero_ws<<<97, 256, 0, stream>>>((uint4*)hbf0, 24704);

  lstm_persist<<<NBLOCKS, 256, LDS_TOTAL, stream>>>(
      xbf, wt, bf_, bb_, hbf0, hbf1, cst, ctr, out);
}
